// GraphSAGE_17678085391128
// MI455X (gfx1250) — matrix operand order repaired and checked by reading, not compiled, not run on hardware
//
#include <hip/hip_runtime.h>

// ---------------------------------------------------------------------------
// GraphSAGE 2-layer forward for MI455X (gfx1250, wave32, WMMA).
//   - scatter phase: HBM/L2-atomic bound (~3.3 GB total, agg fits in 192MB L2)
//   - GEMM phase: v_wmma_f32_16x16x32_bf16 with bf16 hi/lo 3-product split
//     (fp32-level accuracy).  Weights pre-swizzled into WMMA B-fragment
//     layout so each lane loads its fragment as 2x global_load_b128.
//   - 32 output rows per block; each wave holds two 16x16 accumulators that
//     share B fragments (2x arithmetic intensity on the weight stream).
// ---------------------------------------------------------------------------

typedef __attribute__((ext_vector_type(16))) __bf16 v16bf;
typedef __attribute__((ext_vector_type(8)))  float  v8f;

#define N_NODES 100000
#define N_EDGES 1600000
#define DIM     128
#define MROWS   32          // rows per GEMM block (100000 = 3125 * 32)
#define WSZ     (DIM * DIM) // elements per weight matrix

// ---------------- utility kernels ----------------

__global__ void sage_zero_f32(float* __restrict__ p, int n) {
    int i = blockIdx.x * 256 + threadIdx.x;
    int stride = gridDim.x * 256;
    for (; i < n; i += stride) p[i] = 0.0f;
}

// float -> bf16 hi/lo split, stored in WMMA B-fragment order:
//   o = ((kc*8 + ntile)*32 + lane)*16 + e
//   maps to W[k = kc*32 + (lane>>4)*16 + e][n = ntile*16 + (lane&15)]
__global__ void sage_cvt_w_swizzle(const float* __restrict__ w,
                                   __bf16* __restrict__ hi,
                                   __bf16* __restrict__ lo) {
    int o = blockIdx.x * 256 + threadIdx.x;
    if (o >= WSZ) return;
    int e    = o & 15;
    int lane = (o >> 4) & 31;
    int nt   = (o >> 9) & 7;
    int kc   = o >> 12;
    int k = kc * 32 + (lane >> 4) * 16 + e;
    int n = nt * 16 + (lane & 15);
    float v = w[k * DIM + n];
    __bf16 h = (__bf16)v;
    hi[o] = h;
    lo[o] = (__bf16)(v - (float)h);
}

__global__ void sage_degree(const int* __restrict__ dst, float* __restrict__ deg) {
    int e = blockIdx.x * 256 + threadIdx.x;
    if (e < N_EDGES) atomicAdd(&deg[dst[e]], 1.0f);
}

__global__ void sage_invdeg(float* __restrict__ deg) {
    int i = blockIdx.x * 256 + threadIdx.x;
    if (i < N_NODES) {
        float d = deg[i];
        deg[i] = (d > 0.0f) ? (1.0f / d) : 0.0f;   // becomes deg_inv in place
    }
}

// one 32-lane wave per edge: gather feat[src] (float4/lane) and atomically
// accumulate into agg[dst].  Coalesced 512B reads; f32 atomics land in L2.
__global__ void sage_scatter(const float* __restrict__ feat,
                             const int* __restrict__ src,
                             const int* __restrict__ dst,
                             float* __restrict__ agg) {
    unsigned tid = blockIdx.x * 256u + threadIdx.x;
    unsigned e = tid >> 5;
    if (e >= N_EDGES) return;
    unsigned lane = tid & 31u;
    int s = src[e], d = dst[e];
    const float4 v = ((const float4*)(feat + (size_t)s * DIM))[lane];
    float* base = agg + (size_t)d * DIM + lane * 4;
    atomicAdd(base + 0, v.x);
    atomicAdd(base + 1, v.y);
    atomicAdd(base + 2, v.z);
    atomicAdd(base + 3, v.w);
}

// ---------------- fused SAGE GEMM: out = act((agg*deg_inv)@Wl + x@Wr + b) ---
// block = 8 waves, 32 rows; wave w -> output cols 16w..16w+15, two M-tiles.

__global__ __launch_bounds__(256) void sage_gemm(
    const float* __restrict__ agg, const float* __restrict__ deg_inv,
    const float* __restrict__ xr,
    const __bf16* __restrict__ Wl_hi, const __bf16* __restrict__ Wl_lo,
    const __bf16* __restrict__ Wr_hi, const __bf16* __restrict__ Wr_lo,
    const float* __restrict__ bias, float* __restrict__ out, int relu)
{
    __shared__ __bf16 Ahi[2][MROWS][DIM];
    __shared__ __bf16 Alo[2][MROWS][DIM];

    const int m0 = blockIdx.x * MROWS;
    const int t  = threadIdx.x;

    // cooperative stage of the two 32x128 A tiles (scaled agg, root x)
    for (int i = t; i < MROWS * DIM; i += 256) {
        int r = i >> 7;
        int k = i & (DIM - 1);
        float di = deg_inv[m0 + r];
        float v0 = agg[(size_t)(m0 + r) * DIM + k] * di;
        __bf16 h0 = (__bf16)v0;
        Ahi[0][r][k] = h0;
        Alo[0][r][k] = (__bf16)(v0 - (float)h0);
        float v1 = xr[(size_t)(m0 + r) * DIM + k];
        __bf16 h1 = (__bf16)v1;
        Ahi[1][r][k] = h1;
        Alo[1][r][k] = (__bf16)(v1 - (float)h1);
    }
    __syncthreads();

    const int lane = t & 31;
    const int half = lane >> 4;      // 0: lanes 0-15, 1: lanes 16-31
    const int lm   = lane & 15;
    const int wave = t >> 5;
    const int n0   = wave * 16;

    v8f acc0 = {};   // rows m0 .. m0+15
    v8f acc1 = {};   // rows m0+16 .. m0+31

    #pragma unroll
    for (int s = 0; s < 2; ++s) {
        const __bf16* __restrict__ Bh = (s == 0) ? Wl_hi : Wr_hi;
        const __bf16* __restrict__ Bl = (s == 0) ? Wl_lo : Wr_lo;
        #pragma unroll
        for (int kc = 0; kc < DIM / 32; ++kc) {
            const int kbase = kc * 32;
            // B fragments: pre-swizzled, contiguous 32B per lane
            const size_t bo = (size_t)(((kc * 8 + wave) * 32 + lane)) * 16;
            v16bf b_hi = *(const v16bf*)(Bh + bo);
            v16bf b_lo = *(const v16bf*)(Bl + bo);

            // A fragments for both M-tiles: k = kbase + half*16 + e (matches B swizzle)
            v16bf a0h, a0l, a1h, a1l;
            #pragma unroll
            for (int e = 0; e < 16; ++e) {
                int k = kbase + half * 16 + e;
                a0h[e] = Ahi[s][lm][k];
                a0l[e] = Alo[s][lm][k];
                a1h[e] = Ahi[s][16 + lm][k];
                a1l[e] = Alo[s][16 + lm][k];
            }
            acc0 = __builtin_amdgcn_wmma_f32_16x16x32_bf16(
                false, a0h, false, b_hi, (short)0, acc0, false, false);
            acc0 = __builtin_amdgcn_wmma_f32_16x16x32_bf16(
                false, a0h, false, b_lo, (short)0, acc0, false, false);
            acc0 = __builtin_amdgcn_wmma_f32_16x16x32_bf16(
                false, a0l, false, b_hi, (short)0, acc0, false, false);
            acc1 = __builtin_amdgcn_wmma_f32_16x16x32_bf16(
                false, a1h, false, b_hi, (short)0, acc1, false, false);
            acc1 = __builtin_amdgcn_wmma_f32_16x16x32_bf16(
                false, a1h, false, b_lo, (short)0, acc1, false, false);
            acc1 = __builtin_amdgcn_wmma_f32_16x16x32_bf16(
                false, a1l, false, b_hi, (short)0, acc1, false, false);
        }
    }

    const float bn = bias[n0 + lm];
    #pragma unroll
    for (int r = 0; r < 8; ++r) {
        float v = acc0[r] + bn;
        if (relu) v = fmaxf(v, 0.0f);
        int m = m0 + half * 8 + r;   // ISA 32-bit C/D layout
        out[(size_t)m * DIM + n0 + lm] = v;
    }
    #pragma unroll
    for (int r = 0; r < 8; ++r) {
        float v = acc1[r] + bn;
        if (relu) v = fmaxf(v, 0.0f);
        int m = m0 + 16 + half * 8 + r;
        out[(size_t)m * DIM + n0 + lm] = v;
    }
}

// ---------------- host-side launch ----------------

extern "C" void kernel_launch(void* const* d_in, const int* in_sizes, int n_in,
                              void* d_out, int out_size, void* d_ws, size_t ws_size,
                              hipStream_t stream) {
    const float* x   = (const float*)d_in[0];
    const int*   ei  = (const int*)d_in[1];   // [2, E]
    const float* W1l = (const float*)d_in[2];
    const float* b1  = (const float*)d_in[3];
    const float* W1r = (const float*)d_in[4];
    const float* W2l = (const float*)d_in[5];
    const float* b2  = (const float*)d_in[6];
    const float* W2r = (const float*)d_in[7];
    float* out = (float*)d_out;

    const int* src = ei;
    const int* dst = ei + N_EDGES;

    // workspace layout
    char* wsp = (char*)d_ws;
    float* deg = (float*)wsp;                                   // N (-> deg_inv)
    float* agg = (float*)(wsp + (((size_t)N_NODES * 4 + 255) & ~(size_t)255));
    float* h   = agg + (size_t)N_NODES * DIM;
    __bf16* wb = (__bf16*)(h + (size_t)N_NODES * DIM);          // 32B aligned
    __bf16* W1l_hi = wb + 0 * WSZ; __bf16* W1l_lo = wb + 1 * WSZ;
    __bf16* W1r_hi = wb + 2 * WSZ; __bf16* W1r_lo = wb + 3 * WSZ;
    __bf16* W2l_hi = wb + 4 * WSZ; __bf16* W2l_lo = wb + 5 * WSZ;
    __bf16* W2r_hi = wb + 6 * WSZ; __bf16* W2r_lo = wb + 7 * WSZ;

    const int ZB = 1024;  // grid-stride zero blocks
    sage_zero_f32<<<ZB, 256, 0, stream>>>(deg, N_NODES);
    sage_zero_f32<<<ZB, 256, 0, stream>>>(agg, N_NODES * DIM);

    sage_cvt_w_swizzle<<<WSZ / 256, 256, 0, stream>>>(W1l, W1l_hi, W1l_lo);
    sage_cvt_w_swizzle<<<WSZ / 256, 256, 0, stream>>>(W1r, W1r_hi, W1r_lo);
    sage_cvt_w_swizzle<<<WSZ / 256, 256, 0, stream>>>(W2l, W2l_hi, W2l_lo);
    sage_cvt_w_swizzle<<<WSZ / 256, 256, 0, stream>>>(W2r, W2r_hi, W2r_lo);

    sage_degree<<<N_EDGES / 256, 256, 0, stream>>>(dst, deg);
    sage_invdeg<<<(N_NODES + 255) / 256, 256, 0, stream>>>(deg);

    // layer 1
    sage_scatter<<<(N_EDGES * 32u) / 256, 256, 0, stream>>>(x, src, dst, agg);
    sage_gemm<<<N_NODES / MROWS, 256, 0, stream>>>(agg, deg, x,
        W1l_hi, W1l_lo, W1r_hi, W1r_lo, b1, h, 1);

    // layer 2
    sage_zero_f32<<<ZB, 256, 0, stream>>>(agg, N_NODES * DIM);
    sage_scatter<<<(N_EDGES * 32u) / 256, 256, 0, stream>>>(h, src, dst, agg);
    sage_gemm<<<N_NODES / MROWS, 256, 0, stream>>>(agg, deg, h,
        W2l_hi, W2l_lo, W2r_hi, W2r_lo, b2, out, 0);
}
